// Decoder_59407987638857
// MI455X (gfx1250) — matrix operand order repaired and checked by reading, not compiled, not run on hardware
//
#include <hip/hip_runtime.h>
#include <hip/hip_bf16.h>
#include <math.h>

// ---------------------------------------------------------------------------
// Tacotron2-style decoder for MI455X (gfx1250, wave32, WMMA).
//
// Strategy:
//  * All four LSTM weight matrices are packed once per call into f16,
//    pre-transposed into WMMA-B fragment order (42 MB -> L2 resident).
//  * Prenet + f0 prenet precomputed for all 800 steps (teacher forcing).
//  * processed_memory precomputed.
//  * One persistent kernel (64 WGs x 256 thr = 8 waves) runs the 800-step
//    scan with global sense-reversing barriers between phases. LSTM GEMMs
//    use v_wmma_f32_16x16x32_f16 with A staged fragment-packed in LDS
//    (up to 188 KB dynamic LDS -- CDNA5 WGP has 320 KB).
//
// Workspace requirement: ~78 MB.
// ---------------------------------------------------------------------------

typedef __attribute__((ext_vector_type(16))) _Float16 v16h;
typedef __attribute__((ext_vector_type(8)))  float    v8f;

#define NWG     64
#define NTHR    256
#define Bb      32
#define T_IN    400
#define T_OUT   800
#define N_MEL   80
#define E_DIM   896
#define H_DIM   1024
#define PRE     256
#define A_DIM   128
#define KT_ATT  69   // 288 (xt padded 257->288) + 896 (ctx) + 1024 (h) = 2208 = 69*32
#define KT_DEC  92   // 1024 (h_att) + 896 (ctx) + 1024 (h_dec) = 2944 = 92*32

// ---------------- workspace layout (float offsets) -------------------------
#define WS_HA      0u
#define WS_CA      32768u
#define WS_HD      65536u
#define WS_CD      98304u
#define WS_CTX     131072u            // 32*896
#define WS_ATTW    159744u            // 32*400
#define WS_ATTC    172544u            // 32*400
#define WS_BAR     185344u            // 2 x u32 (+pad)
#define WS_PQ      185352u            // 32*128
#define WS_STEPIN  189448u            // 800*32*257
#define WS_PM      6768648u           // 32*400*128
#define WS_LOC     8407048u           // 32*400*32
#define WS_F16     8816648u           // f16 region start (32B aligned)
#define NHALF_BATT ((size_t)256 * KT_ATT * 512)   // 9,043,968 halves
#define NHALF_BDEC ((size_t)256 * KT_DEC * 512)   // 12,058,624 halves
#define WS_ZERO_N  185352u            // floats to zero (state + barrier)

__device__ __forceinline__ float sigm(float x) { return 1.0f / (1.0f + __expf(-x)); }

// ---------------------------------------------------------------------------
// grid-wide sense-reversing barrier (persistent kernel)
// ---------------------------------------------------------------------------
__device__ __forceinline__ void gsync(unsigned* bar, unsigned* gen, unsigned& mygen) {
  __syncthreads();
  if (threadIdx.x == 0) {
    unsigned g = mygen;
    unsigned prev = __hip_atomic_fetch_add(bar, 1u, __ATOMIC_ACQ_REL, __HIP_MEMORY_SCOPE_AGENT);
    if (prev == NWG - 1) {
      __hip_atomic_store(bar, 0u, __ATOMIC_RELAXED, __HIP_MEMORY_SCOPE_AGENT);
      __hip_atomic_fetch_add(gen, 1u, __ATOMIC_RELEASE, __HIP_MEMORY_SCOPE_AGENT);
    } else {
      while (__hip_atomic_load(gen, __ATOMIC_ACQUIRE, __HIP_MEMORY_SCOPE_AGENT) == g) {
        __builtin_amdgcn_s_sleep(2);
      }
    }
    mygen = g + 1;
  }
  __syncthreads();
}

// ---------------------------------------------------------------------------
// build f16 A-matrix in LDS, fragment-packed:
//   aS[(kt*2+mt)*512 + lane*16 + e] where
//   m = mt*16 + (lane&15)
//   k = kt*32 + (e&1) + 2*((e>>1)&3) + (e>=8?16:0) + (lane>=16?8:0)
// (matches the 16-bit 16x32 A VGPR layout, ISA 7.12.2)
// ---------------------------------------------------------------------------
template <int KT, bool IS_DEC>
__device__ void build_A(_Float16* aS, int t, const float* step_in,
                        const float* ctx, const float* hA, const float* hD) {
  for (int s = threadIdx.x; s < KT * 1024; s += NTHR) {
    int kt   = s >> 10;
    int r    = s & 1023;
    int mt   = r >> 9;
    int sl   = r & 511;
    int lane = sl >> 4, e = sl & 15;
    int m = mt * 16 + (lane & 15);
    int k = kt * 32 + (e & 1) + (((e >> 1) & 3) << 1) + ((e >= 8) ? 16 : 0) +
            ((lane >= 16) ? 8 : 0);
    float v;
    if (!IS_DEC) {
      if (k < 288)        v = (k < 257) ? step_in[(size_t)(t * Bb + m) * 257 + k] : 0.0f;
      else if (k < 1184)  v = ctx[m * E_DIM + (k - 288)];
      else                v = hA[m * H_DIM + (k - 1184)];
    } else {
      if (k < 1024)       v = hA[m * H_DIM + k];
      else if (k < 1920)  v = ctx[m * E_DIM + (k - 1024)];
      else                v = hD[m * H_DIM + (k - 1920)];
    }
    aS[s] = (_Float16)v;
  }
}

// ---------------------------------------------------------------------------
// LSTM GEMM + pointwise. 8 waves: wave = (g<<1)|mt, g = gate group (i,f,g,o),
// mt = M tile (batch rows 0-15 / 16-31). Each WG owns 16 hidden columns.
// ---------------------------------------------------------------------------
template <int KT>
__device__ void lstm_gemm(const _Float16* aS, const _Float16* Bp,
                          const float* bih, const float* bhh,
                          float* hOut, float* cOut, float* gbuf, int wg) {
  const int tid = threadIdx.x, lane = tid & 31, wv = tid >> 5;
  const int mt = wv & 1, g = wv >> 1;
  v8f acc = {};
  const _Float16* aBase = aS + (size_t)mt * 512 + (size_t)lane * 16;
  const int ntile = g * 64 + wg;
  const _Float16* bBase = Bp + (size_t)ntile * KT * 512 + (size_t)lane * 16;
  for (int kt = 0; kt < KT; ++kt) {
    v16h a = *(const v16h*)(aBase + (size_t)kt * 1024);
    v16h b = *(const v16h*)(bBase + (size_t)kt * 512);
    if (kt + 1 < KT) __builtin_prefetch(bBase + (size_t)(kt + 1) * 512, 0, 1);
    acc = __builtin_amdgcn_wmma_f32_16x16x32_f16(false, a, false, b,
                                                 (short)0, acc, false, false);
  }
  __syncthreads();  // everyone done reading aS; reuse LDS for gate exchange
  const int c = lane & 15;
  const int n = g * 1024 + wg * 16 + c;
  const float bias = bih[n] + bhh[n];
#pragma unroll
  for (int r = 0; r < 8; ++r) {
    int brow = mt * 16 + r + ((lane >= 16) ? 8 : 0);
    gbuf[(g * 32 + brow) * 16 + c] = acc[r] + bias;
  }
  __syncthreads();
#pragma unroll
  for (int it = 0; it < 2; ++it) {
    int idx = tid + it * NTHR;
    int b = idx >> 4, cc = idx & 15;
    int hid = wg * 16 + cc;
    float iv = gbuf[(0 * 32 + b) * 16 + cc];
    float fv = gbuf[(1 * 32 + b) * 16 + cc];
    float gv = gbuf[(2 * 32 + b) * 16 + cc];
    float ov = gbuf[(3 * 32 + b) * 16 + cc];
    float cp = cOut[b * H_DIM + hid];
    float cn = sigm(fv) * cp + sigm(iv) * tanhf(gv);
    cOut[b * H_DIM + hid] = cn;
    hOut[b * H_DIM + hid] = sigm(ov) * tanhf(cn);
  }
}

// ---------------------------------------------------------------------------
// prep kernels
// ---------------------------------------------------------------------------
__global__ void zero_kernel(float* p, unsigned n) {
  size_t i = (size_t)blockIdx.x * blockDim.x + threadIdx.x;
  size_t st = (size_t)gridDim.x * blockDim.x;
  for (; i < n; i += st) p[i] = 0.0f;
}

// pack concat(wih[:, :257] pad 288, wih[:, 257:], whh) into B-fragment order
__global__ void pack_att_kernel(const float* wih, const float* whh, _Float16* out) {
  size_t idx = (size_t)blockIdx.x * blockDim.x + threadIdx.x;
  size_t total = NHALF_BATT;
  size_t st = (size_t)gridDim.x * blockDim.x;
  for (; idx < total; idx += st) {
    int s = (int)(idx & 511);
    size_t q = idx >> 9;
    int kt = (int)(q % KT_ATT);
    int ntile = (int)(q / KT_ATT);
    int lane = s >> 4, e = s & 15;
    int n = ntile * 16 + (lane & 15);
    int k = kt * 32 + (e & 7) + ((e >= 8) ? 16 : 0) + ((lane >= 16) ? 8 : 0);  // B 32x16 layout (ISA 7.12.4)
    float v;
    if (k < 288)       v = (k < 257) ? wih[(size_t)n * 1153 + k] : 0.0f;
    else if (k < 1184) v = wih[(size_t)n * 1153 + 257 + (k - 288)];
    else               v = whh[(size_t)n * 1024 + (k - 1184)];
    out[idx] = (_Float16)v;
  }
}

__global__ void pack_dec_kernel(const float* wih, const float* whh, _Float16* out) {
  size_t idx = (size_t)blockIdx.x * blockDim.x + threadIdx.x;
  size_t total = NHALF_BDEC;
  size_t st = (size_t)gridDim.x * blockDim.x;
  for (; idx < total; idx += st) {
    int s = (int)(idx & 511);
    size_t q = idx >> 9;
    int kt = (int)(q % KT_DEC);
    int ntile = (int)(q / KT_DEC);
    int lane = s >> 4, e = s & 15;
    int n = ntile * 16 + (lane & 15);
    int k = kt * 32 + (e & 7) + ((e >= 8) ? 16 : 0) + ((lane >= 16) ? 8 : 0);
    float v;
    if (k < 1920) v = wih[(size_t)n * 1920 + k];
    else          v = whh[(size_t)n * 1024 + (k - 1920)];
    out[idx] = (_Float16)v;
  }
}

// prenet (2x relu MLP) + f0 pointwise -> step_in[t][b][257]
__global__ void prenet_kernel(const float* dec_inputs, const float* f0s,
                              const float* w1, const float* w2, const float* pf0,
                              float* step_in) {
  int t = blockIdx.x >> 5;
  int b = blockIdx.x & 31;
  __shared__ float s_in[N_MEL];
  __shared__ float s_h1[PRE];
  int tid = threadIdx.x;
  if (tid < N_MEL)
    s_in[tid] = (t == 0) ? 0.0f
                         : dec_inputs[((size_t)b * N_MEL + tid) * T_OUT + (t - 1)];
  __syncthreads();
  float a = 0.0f;
  for (int m = 0; m < N_MEL; ++m) a += s_in[m] * w1[tid * N_MEL + m];
  s_h1[tid] = fmaxf(a, 0.0f);
  __syncthreads();
  float o = 0.0f;
  for (int p = 0; p < PRE; ++p) o += s_h1[p] * w2[tid * PRE + p];
  step_in[((size_t)t * Bb + b) * 257 + tid] = fmaxf(o, 0.0f);
  if (tid == 0) {
    float f = f0s[(size_t)b * T_OUT + t] * pf0[0];
    step_in[((size_t)t * Bb + b) * 257 + 256] = fmaxf(f, 0.0f);
  }
}

// processed_memory[b][t][a] = memory[b][t][:] . memory_w[a][:]
__global__ void pm_kernel(const float* memory, const float* memw, float* pm) {
  int bt = blockIdx.x;  // 0..12799
  __shared__ float srow[E_DIM];
  int tid = threadIdx.x;  // 128
  for (int i = tid; i < E_DIM; i += 128) srow[i] = memory[(size_t)bt * E_DIM + i];
  __syncthreads();
  float acc = 0.0f;
  for (int e = 0; e < E_DIM; ++e) acc += srow[e] * memw[tid * E_DIM + e];
  pm[(size_t)bt * A_DIM + tid] = acc;
}

// ---------------------------------------------------------------------------
// persistent decoder scan
// ---------------------------------------------------------------------------
__global__ void __launch_bounds__(NTHR)
decoder_persistent(const float* step_in, const float* memory, const int* memlen,
                   const float* pm, const _Float16* Batt, const _Float16* Bdec,
                   const float* att_bih, const float* att_bhh,
                   const float* dec_bih, const float* dec_bhh,
                   const float* query_w, const float* v_w,
                   const float* loc_conv_w, const float* loc_dense_w,
                   const float* proj_w, const float* proj_b,
                   const float* gate_w, const float* gate_b,
                   float* hA, float* cA, float* hD, float* cD,
                   float* ctx, float* attw, float* attcum, float* pq,
                   float* locbuf, unsigned* bar, unsigned* gen,
                   float* out_mel, float* out_gate, float* out_align) {
  extern __shared__ __align__(16) unsigned char smem_raw[];
  _Float16* aS = (_Float16*)smem_raw;
  float* fS = (float*)smem_raw;
  const int wg = blockIdx.x, tid = threadIdx.x;
  unsigned mygen = 0;

  for (int t = 0; t < T_OUT; ++t) {
    // ---------------- Phase A: attention LSTM ----------------------------
    build_A<KT_ATT, false>(aS, t, step_in, ctx, hA, hD);
    gsync(bar, gen, mygen);  // all builds read hA/ctx before hA is rewritten
    lstm_gemm<KT_ATT>(aS, Batt, att_bih, att_bhh, hA, cA, fS, wg);
    gsync(bar, gen, mygen);

    // ---------------- Phase B: location conv (wg<32) + pq (wg>=32) -------
    if (wg < Bb) {
      const int b = wg;
      float* sw  = fS;            // att_w row     [400]
      float* sc  = fS + 400;      // att_w_cum row [400]
      float* swt = fS + 800;      // conv weights  [32*2*31]
      for (int i = tid; i < T_IN; i += NTHR) {
        sw[i] = attw[b * T_IN + i];
        sc[i] = attcum[b * T_IN + i];
      }
      for (int i = tid; i < 32 * 2 * 31; i += NTHR) swt[i] = loc_conv_w[i];
      __syncthreads();
      for (int idx = tid; idx < T_IN * 32; idx += NTHR) {
        int tp = idx >> 5, f = idx & 31;
        float acc = 0.0f;
#pragma unroll
        for (int k = 0; k < 31; ++k) {
          int p = tp + k - 15;
          if (p >= 0 && p < T_IN)
            acc += sw[p] * swt[(f * 2 + 0) * 31 + k] +
                   sc[p] * swt[(f * 2 + 1) * 31 + k];
        }
        locbuf[((size_t)b * T_IN + tp) * 32 + f] = acc;
      }
    } else {
      int a0 = (wg - Bb) * 4;
      if (tid < 128) {
        int b = tid >> 2, a = a0 + (tid & 3);
        float acc = 0.0f;
        for (int k = 0; k < H_DIM; ++k) acc += hA[b * H_DIM + k] * query_w[a * H_DIM + k];
        pq[b * A_DIM + a] = acc;
      }
    }
    gsync(bar, gen, mygen);

    // ---------------- Phase C: energies / softmax / context (wg<32) ------
    if (wg < Bb) {
      const int b = wg;
      float* sE  = fS;             // [400]
      float* sPQ = fS + 400;       // [128]
      float* sV  = fS + 528;       // [128]
      float* sLD = fS + 656;       // [128*32]
      float* sR  = fS + 4752;      // [256]
      for (int i = tid; i < A_DIM; i += NTHR) { sPQ[i] = pq[b * A_DIM + i]; sV[i] = v_w[i]; }
      for (int i = tid; i < A_DIM * 32; i += NTHR) sLD[i] = loc_dense_w[i];
      __syncthreads();
      int ml = memlen[b];
      for (int tp = tid; tp < T_IN; tp += NTHR) {
        float lrow[32];
#pragma unroll
        for (int f = 0; f < 32; ++f) lrow[f] = locbuf[((size_t)b * T_IN + tp) * 32 + f];
        float en = 0.0f;
        for (int a = 0; a < A_DIM; ++a) {
          float l = 0.0f;
#pragma unroll
          for (int f = 0; f < 32; ++f) l += lrow[f] * sLD[a * 32 + f];
          float x = sPQ[a] + l + pm[((size_t)b * T_IN + tp) * A_DIM + a];
          en += sV[a] * tanhf(x);
        }
        sE[tp] = (tp >= ml) ? -1e30f : en;
      }
      __syncthreads();
      float m = -1e30f;
      for (int i = tid; i < T_IN; i += NTHR) m = fmaxf(m, sE[i]);
      sR[tid] = m;
      __syncthreads();
      for (int s = 128; s > 0; s >>= 1) {
        if (tid < s) sR[tid] = fmaxf(sR[tid], sR[tid + s]);
        __syncthreads();
      }
      float mx = sR[0];
      __syncthreads();
      float ssum = 0.0f;
      for (int i = tid; i < T_IN; i += NTHR) {
        float e = __expf(sE[i] - mx);
        sE[i] = e;
        ssum += e;
      }
      __syncthreads();
      sR[tid] = ssum;
      __syncthreads();
      for (int s = 128; s > 0; s >>= 1) {
        if (tid < s) sR[tid] += sR[tid + s];
        __syncthreads();
      }
      float inv = 1.0f / sR[0];
      __syncthreads();
      for (int i = tid; i < T_IN; i += NTHR) {
        float w = sE[i] * inv;
        sE[i] = w;
        attw[b * T_IN + i] = w;
        attcum[b * T_IN + i] += w;
        out_align[((size_t)b * T_OUT + t) * T_IN + i] = w;
      }
      __syncthreads();
      for (int e = tid; e < E_DIM; e += NTHR) {
        float acc = 0.0f;
        for (int tp = 0; tp < T_IN; ++tp)
          acc += sE[tp] * memory[((size_t)b * T_IN + tp) * E_DIM + e];
        ctx[b * E_DIM + e] = acc;
      }
    }
    gsync(bar, gen, mygen);

    // ---------------- Phase D: decoder LSTM ------------------------------
    build_A<KT_DEC, true>(aS, t, step_in, ctx, hA, hD);
    gsync(bar, gen, mygen);  // builds read hD before hD is rewritten
    lstm_gemm<KT_DEC>(aS, Bdec, dec_bih, dec_bhh, hD, cD, fS, wg);
    gsync(bar, gen, mygen);

    // ---------------- Phase E: mel / gate projections (wg<32) ------------
    if (wg < Bb) {
      const int b = wg;
      float* shc = fS;  // [1920]
      for (int i = tid; i < H_DIM; i += NTHR) shc[i] = hD[b * H_DIM + i];
      for (int i = tid; i < E_DIM; i += NTHR) shc[H_DIM + i] = ctx[b * E_DIM + i];
      __syncthreads();
      if (tid < N_MEL) {
        float acc = proj_b[tid];
        for (int k = 0; k < 1920; ++k) acc += shc[k] * proj_w[tid * 1920 + k];
        out_mel[((size_t)b * N_MEL + tid) * T_OUT + t] = acc;
      } else if (tid == N_MEL) {
        float acc = gate_b[0];
        for (int k = 0; k < 1920; ++k) acc += shc[k] * gate_w[k];
        out_gate[(size_t)b * T_OUT + t] = acc;
      }
    }
    gsync(bar, gen, mygen);
  }
}

// ---------------------------------------------------------------------------
extern "C" void kernel_launch(void* const* d_in, const int* in_sizes, int n_in,
                              void* d_out, int out_size, void* d_ws, size_t ws_size,
                              hipStream_t stream) {
  const float* memory      = (const float*)d_in[0];
  const float* dec_inputs  = (const float*)d_in[1];
  const float* f0s         = (const float*)d_in[2];
  const int*   memlen      = (const int*)  d_in[3];
  const float* prenet_w1   = (const float*)d_in[4];
  const float* prenet_w2   = (const float*)d_in[5];
  const float* prenet_f0_w = (const float*)d_in[6];
  const float* att_wih     = (const float*)d_in[7];
  const float* att_whh     = (const float*)d_in[8];
  const float* att_bih     = (const float*)d_in[9];
  const float* att_bhh     = (const float*)d_in[10];
  const float* query_w     = (const float*)d_in[11];
  const float* memory_w    = (const float*)d_in[12];
  const float* v_w         = (const float*)d_in[13];
  const float* loc_conv_w  = (const float*)d_in[14];
  const float* loc_dense_w = (const float*)d_in[15];
  const float* dec_wih     = (const float*)d_in[16];
  const float* dec_whh     = (const float*)d_in[17];
  const float* dec_bih     = (const float*)d_in[18];
  const float* dec_bhh     = (const float*)d_in[19];
  const float* proj_w      = (const float*)d_in[20];
  const float* proj_b      = (const float*)d_in[21];
  const float* gate_w      = (const float*)d_in[22];
  const float* gate_b      = (const float*)d_in[23];

  float* ws = (float*)d_ws;
  float* hA      = ws + WS_HA;
  float* cA      = ws + WS_CA;
  float* hD      = ws + WS_HD;
  float* cD      = ws + WS_CD;
  float* ctx     = ws + WS_CTX;
  float* attw    = ws + WS_ATTW;
  float* attcum  = ws + WS_ATTC;
  unsigned* bar  = (unsigned*)(ws + WS_BAR);
  unsigned* gen  = bar + 1;
  float* pq      = ws + WS_PQ;
  float* step_in = ws + WS_STEPIN;
  float* pm      = ws + WS_PM;
  float* locbuf  = ws + WS_LOC;
  _Float16* Batt = (_Float16*)(ws + WS_F16);
  _Float16* Bdec = Batt + NHALF_BATT;

  float* out_mel   = (float*)d_out;
  float* out_gate  = out_mel + (size_t)Bb * N_MEL * T_OUT;   // 2,048,000
  float* out_align = out_gate + (size_t)Bb * T_OUT;          // +25,600

  // prep: zero state + barrier, pack f16 weights, prenet, processed_memory
  zero_kernel<<<512, 256, 0, stream>>>(ws, WS_ZERO_N);
  pack_att_kernel<<<4096, 256, 0, stream>>>(att_wih, att_whh, Batt);
  pack_dec_kernel<<<4096, 256, 0, stream>>>(dec_wih, dec_whh, Bdec);
  prenet_kernel<<<T_OUT * Bb, 256, 0, stream>>>(dec_inputs, f0s, prenet_w1,
                                                prenet_w2, prenet_f0_w, step_in);
  pm_kernel<<<Bb * T_IN, 128, 0, stream>>>(memory, memory_w, pm);

  // persistent scan: 64 WGs, 8 waves each, 188416 B dynamic LDS (dec A frags)
  size_t smem = (size_t)KT_DEC * 1024 * sizeof(_Float16);  // 188,416 bytes
  decoder_persistent<<<NWG, NTHR, smem, stream>>>(
      step_in, memory, memlen, pm, Batt, Bdec,
      att_bih, att_bhh, dec_bih, dec_bhh,
      query_w, v_w, loc_conv_w, loc_dense_w,
      proj_w, proj_b, gate_w, gate_b,
      hA, cA, hD, cD, ctx, attw, attcum, pq, locbuf,
      bar, gen, out_mel, out_gate, out_align);
}
